// SelfAttention_5514738008241
// MI455X (gfx1250) — matrix operand order repaired and checked by reading, not compiled, not run on hardware
//
#include <hip/hip_runtime.h>
#include <hip/hip_bf16.h>

// ---------------- types ----------------
typedef _Float16 v8h  __attribute__((ext_vector_type(8)));
typedef _Float16 v16h __attribute__((ext_vector_type(16)));
typedef float    v8f  __attribute__((ext_vector_type(8)));

union frag16 { v16h v; v8h h[2]; };

static __device__ __forceinline__ v8f wmma_f16(v16h a, v16h b, v8f c) {
    // D = A(16x32 f16) * B(32x16 f16) + C(16x16 f32)
    return __builtin_amdgcn_wmma_f32_16x16x32_f16(
        /*neg_a=*/false, a, /*neg_b=*/false, b,
        /*c_mod=*/(short)0, c, /*reuse_a=*/false, /*reuse_b=*/false);
}

static __device__ __forceinline__ float rmax16(float v) {
#pragma unroll
    for (int m = 8; m >= 1; m >>= 1) v = fmaxf(v, __shfl_xor(v, m, 32));
    return v;
}
static __device__ __forceinline__ float rsum16(float v) {
#pragma unroll
    for (int m = 8; m >= 1; m >>= 1) v += __shfl_xor(v, m, 32);
    return v;
}

// LDS byte offset = low 32 bits of the generic pointer
static __device__ __forceinline__ unsigned lds_off(const void* p) {
    return (unsigned)(unsigned long long)p;
}

// CDNA5 async copy: global -> LDS, 16 bytes per lane, tracked by ASYNCcnt.
static __device__ __forceinline__ void async_b128_to_lds(unsigned lds_addr,
                                                         const void* gaddr) {
    asm volatile("global_load_async_to_lds_b128 %0, %1, off"
                 :: "v"(lds_addr), "v"(gaddr) : "memory");
}
static __device__ __forceinline__ void wait_asynccnt0() {
    asm volatile("s_wait_asynccnt 0x0" ::: "memory");
}

// ---------------- problem constants ----------------
#define BATCH   2
#define SEQ     2048
#define EMB     1024
#define NHEAD   16
#define HDIM    64
#define NTOK    (BATCH * SEQ)          // 4096

// ---------------- conversion kernels ----------------
__global__ void __launch_bounds__(256)
cvt_f32_to_f16(const float* __restrict__ in, _Float16* __restrict__ out, int n) {
    int i = blockIdx.x * blockDim.x + threadIdx.x;
    if (i < n) out[i] = (_Float16)in[i];
}

// in [R][C] f32  ->  out [C][R] f16   (weight transpose for B-fragment loads)
__global__ void __launch_bounds__(256)
transpose_cvt_f16(const float* __restrict__ in, _Float16* __restrict__ out,
                  int R, int C) {
    int o = blockIdx.x * blockDim.x + threadIdx.x;
    if (o >= R * C) return;
    int c = o / R;          // output row  == input col
    int r = o % R;          // output col  == input row
    out[o] = (_Float16)in[(size_t)r * C + c];
}

// v_h [B*S][EMB] f16  ->  v_t [B*H][HDIM][S] f16  (V^T per head, contiguous in s)
__global__ void __launch_bounds__(256)
transpose_v_kernel(const _Float16* __restrict__ vh, _Float16* __restrict__ vt) {
    int i = blockIdx.x * blockDim.x + threadIdx.x;
    if (i >= NTOK * EMB) return;
    int s  = i % SEQ;
    int r  = i / SEQ;
    int d  = r % HDIM;
    int bh = r / HDIM;
    int b  = bh / NHEAD;
    int h  = bh % NHEAD;
    vt[i] = vh[(size_t)(b * SEQ + s) * EMB + h * HDIM + d];
}

// ---------------- WMMA GEMM: C[M,N] = A[M,K](f16) @ W[K,N] + bias ----------------
// Bt is W^T stored [N][K] f16.  Block = 8 waves; wave w -> rows [row0+16w, +16),
// 4 col tiles of 16 (block tile 128 x 64).  The shared 64x32-half B tile is
// staged into LDS once per block with async-to-LDS copies, double buffered so
// the copy for k-step i+1 overlaps the WMMAs of step i.
#define BSTRIDE 40   // halves per LDS row (80 B: 16B aligned, bank-padded)

__global__ void __launch_bounds__(256)
gemm_bias_wmma(const _Float16* __restrict__ A, const _Float16* __restrict__ Bt,
               const float* __restrict__ bias,
               _Float16* __restrict__ Ch, float* __restrict__ Cf,
               int M, int N, int K, int out_f32) {
    __shared__ _Float16 lds_b[2][64][BSTRIDE];   // 2 stages x 64 rows x 40 halves

    const int tid   = threadIdx.x;
    const int wave  = tid >> 5;
    const int lane  = tid & 31;
    const int arow  = lane & 15;
    const int ahalf = lane >> 4;
    const int kb    = ahalf * 8;

    const int row0 = blockIdx.y * 128 + wave * 16;
    const int col0 = blockIdx.x * 64;

    // async staging assignment: 256 lanes cover 64 rows x 4 x 16B
    const int cr = tid >> 2;        // B-tile row (= output column), 0..63
    const int cp = tid & 3;         // 16B chunk within 64B row
    const _Float16* gsrc = Bt + (size_t)(col0 + cr) * K + cp * 8;
    const unsigned ldst0 = lds_off(&lds_b[0][cr][cp * 8]);
    const unsigned ldst1 = lds_off(&lds_b[1][cr][cp * 8]);

    v8f acc[4];
#pragma unroll
    for (int t = 0; t < 4; ++t)
#pragma unroll
        for (int i = 0; i < 8; ++i) acc[t][i] = 0.0f;

    const _Float16* arp = A + (size_t)(row0 + arow) * K;

    // prologue: stage k0 = 0 into buffer 0
    async_b128_to_lds(ldst0, (const void*)gsrc);

    for (int k0 = 0; k0 < K; k0 += 32) {
        const int s = (k0 >> 5) & 1;

        // own async copies done, then block-wide visibility
        wait_asynccnt0();
        __syncthreads();

        // kick off the next stage while we compute on this one
        if (k0 + 32 < K)
            async_b128_to_lds(s ? ldst0 : ldst1,
                              (const void*)(gsrc + k0 + 32));

        // A fragment: 16x32 f16 direct from global (wave-private rows)
        frag16 af;
        af.h[0] = *(const v8h*)(arp + k0 + ahalf * 16);
        af.h[1] = *(const v8h*)(arp + k0 + ahalf * 16 + 8);
        if (k0 + 64 < K)
            __builtin_prefetch((const void*)(arp + k0 + 64), 0, 1);

        // B fragments from LDS: lane%16 = column, 16 contiguous K-halves/lane
        frag16 b0, b1, b2, b3;
        b0.h[0] = *(const v8h*)&lds_b[s][ 0 + arow][kb + ahalf * 8];
        b0.h[1] = *(const v8h*)&lds_b[s][ 0 + arow][kb + ahalf * 8 + 8];
        b1.h[0] = *(const v8h*)&lds_b[s][16 + arow][kb + ahalf * 8];
        b1.h[1] = *(const v8h*)&lds_b[s][16 + arow][kb + ahalf * 8 + 8];
        b2.h[0] = *(const v8h*)&lds_b[s][32 + arow][kb + ahalf * 8];
        b2.h[1] = *(const v8h*)&lds_b[s][32 + arow][kb + ahalf * 8 + 8];
        b3.h[0] = *(const v8h*)&lds_b[s][48 + arow][kb + ahalf * 8];
        b3.h[1] = *(const v8h*)&lds_b[s][48 + arow][kb + ahalf * 8 + 8];

        acc[0] = wmma_f16(af.v, b0.v, acc[0]);
        acc[1] = wmma_f16(af.v, b1.v, acc[1]);
        acc[2] = wmma_f16(af.v, b2.v, acc[2]);
        acc[3] = wmma_f16(af.v, b3.v, acc[3]);
    }

#pragma unroll
    for (int t = 0; t < 4; ++t) {
        const int col = col0 + t * 16 + arow;
        const float bv = bias[col];
#pragma unroll
        for (int i = 0; i < 8; ++i) {
            const int row = row0 + i + ahalf * 8;   // C layout: VGPR i -> row
            const float val = acc[t][i] + bv;
            if (out_f32) Cf[(size_t)row * N + col] = val;
            else         Ch[(size_t)row * N + col] = (_Float16)val;
        }
    }
}

// ---------------- Flash attention ----------------
// grid = B*H*(S/64) blocks, 128 threads (4 waves).  Wave owns 16 query rows.
// qh,kh: [B*S][EMB] f16 ; vt: [B*H][HDIM][S] f16 ; oh: [B*S][EMB] f16.
__global__ void __launch_bounds__(128)
flash_attn_wmma(const _Float16* __restrict__ qh, const _Float16* __restrict__ kh,
                const _Float16* __restrict__ vt, _Float16* __restrict__ oh) {
    __shared__ _Float16 lds_p[4][16][40];   // 80B row stride: 16B aligned

    const int wave  = threadIdx.x >> 5;
    const int lane  = threadIdx.x & 31;
    const int arow  = lane & 15;
    const int ahalf = lane >> 4;
    const int kb    = ahalf * 8;

    const int nqt = SEQ / 64;
    const int qt  = blockIdx.x % nqt;
    const int bh  = blockIdx.x / nqt;
    const int b   = bh / NHEAD;
    const int h   = bh % NHEAD;
    const int q0  = qt * 64 + wave * 16;

    // Q fragments for d=[0,32) and d=[32,64)
    const _Float16* qrow = qh + (size_t)(b * SEQ + q0 + arow) * EMB + h * HDIM;
    frag16 aq0, aq1;
    aq0.h[0] = *(const v8h*)(qrow + ahalf * 16);
    aq0.h[1] = *(const v8h*)(qrow + ahalf * 16 + 8);
    aq1.h[0] = *(const v8h*)(qrow + 32 + ahalf * 16);
    aq1.h[1] = *(const v8h*)(qrow + 32 + ahalf * 16 + 8);

    v8f acc[4];
#pragma unroll
    for (int t = 0; t < 4; ++t)
#pragma unroll
        for (int i = 0; i < 8; ++i) acc[t][i] = 0.0f;

    float m[8], l[8];
#pragma unroll
    for (int i = 0; i < 8; ++i) { m[i] = -1e30f; l[i] = 0.0f; }

    const _Float16* kbase = kh + (size_t)(b * SEQ) * EMB + h * HDIM;
    const _Float16* vbase = vt + (size_t)bh * HDIM * SEQ;
    const float scale = 0.125f;   // 1/sqrt(64)

    for (int j = 0; j < SEQ; j += 32) {
        // ---- scores: two 16x16 tiles (keys j..j+15 and j+16..j+31) ----
        v8f s0, s1;
#pragma unroll
        for (int i = 0; i < 8; ++i) { s0[i] = 0.0f; s1[i] = 0.0f; }
        {
            const _Float16* kr = kbase + (size_t)(j + arow) * EMB + ahalf * 16;
            v16h b0 = *(const v16h*)(kr);        // d 0..31
            v16h b1 = *(const v16h*)(kr + 32);   // d 32..63
            s0 = wmma_f16(aq0.v, b0, s0);
            s0 = wmma_f16(aq1.v, b1, s0);
        }
        {
            const _Float16* kr = kbase + (size_t)(j + 16 + arow) * EMB + ahalf * 16;
            v16h b0 = *(const v16h*)(kr);
            v16h b1 = *(const v16h*)(kr + 32);
            s1 = wmma_f16(aq0.v, b0, s1);
            s1 = wmma_f16(aq1.v, b1, s1);
        }

        // ---- online softmax (rows are 16-lane groups of one VGPR slot) ----
        float corr[8];
#pragma unroll
        for (int i = 0; i < 8; ++i) {
            const float a0 = s0[i] * scale;
            const float a1 = s1[i] * scale;
            float tmax = rmax16(fmaxf(a0, a1));
            const float mn = fmaxf(m[i], tmax);
            const float p0 = __expf(a0 - mn);
            const float p1 = __expf(a1 - mn);
            const float c  = __expf(m[i] - mn);
            l[i] = l[i] * c + rsum16(p0 + p1);
            m[i] = mn;
            corr[i] = c;
            // C-layout -> LDS tile (row = i + 8*ahalf, col = key-in-tile)
            lds_p[wave][i + ahalf * 8][arow]      = (_Float16)p0;
            lds_p[wave][i + ahalf * 8][arow + 16] = (_Float16)p1;
        }
#pragma unroll
        for (int t = 0; t < 4; ++t)
#pragma unroll
            for (int i = 0; i < 8; ++i) acc[t][i] *= corr[i];

        // ---- P (16x32) as A fragment from LDS, V^T tiles as B fragments ----
        frag16 pa;
        pa.h[0] = *(const v8h*)&lds_p[wave][arow][ahalf * 16];
        pa.h[1] = *(const v8h*)&lds_p[wave][arow][ahalf * 16 + 8];
#pragma unroll
        for (int t = 0; t < 4; ++t) {
            const _Float16* vr =
                vbase + (size_t)(t * 16 + arow) * SEQ + j + ahalf * 16;
            v16h bf = *(const v16h*)vr;          // 16 contiguous keys per lane
            acc[t] = wmma_f16(pa.v, bf, acc[t]);
        }
    }

    // ---- epilogue: normalize, store f16 ----
    float inv[8];
#pragma unroll
    for (int i = 0; i < 8; ++i) inv[i] = 1.0f / l[i];
#pragma unroll
    for (int t = 0; t < 4; ++t) {
        const int col = h * HDIM + t * 16 + arow;
#pragma unroll
        for (int i = 0; i < 8; ++i) {
            const int row = q0 + i + ahalf * 8;
            oh[(size_t)(b * SEQ + row) * EMB + col] = (_Float16)(acc[t][i] * inv[i]);
        }
    }
}

// ---------------- launch ----------------
extern "C" void kernel_launch(void* const* d_in, const int* in_sizes, int n_in,
                              void* d_out, int out_size, void* d_ws, size_t ws_size,
                              hipStream_t stream) {
    (void)in_sizes; (void)n_in; (void)out_size; (void)ws_size;

    const float* x  = (const float*)d_in[0];
    const float* Wq = (const float*)d_in[1];
    const float* bq = (const float*)d_in[2];
    const float* Wk = (const float*)d_in[3];
    const float* bk = (const float*)d_in[4];
    const float* Wv = (const float*)d_in[5];
    const float* bv = (const float*)d_in[6];
    const float* Wo = (const float*)d_in[7];
    const float* bo = (const float*)d_in[8];
    float* out = (float*)d_out;

    char* ws = (char*)d_ws;
    size_t off = 0;
    auto alloc = [&](size_t bytes) -> void* {
        void* p = ws + off;
        off += (bytes + 255) & ~(size_t)255;
        return p;
    };

    const size_t nX = (size_t)NTOK * EMB;      // 4096*1024
    const size_t nW = (size_t)EMB * EMB;       // 1024*1024

    _Float16* x_h    = (_Float16*)alloc(nX * 2);
    _Float16* Wq_t   = (_Float16*)alloc(nW * 2);
    _Float16* Wk_t   = (_Float16*)alloc(nW * 2);
    _Float16* Wv_t   = (_Float16*)alloc(nW * 2);
    _Float16* Wo_t   = (_Float16*)alloc(nW * 2);
    _Float16* q_h    = (_Float16*)alloc(nX * 2);
    _Float16* k_h    = (_Float16*)alloc(nX * 2);
    _Float16* v_h    = (_Float16*)alloc(nX * 2);
    _Float16* v_t    = (_Float16*)alloc(nX * 2);
    _Float16* attn_h = (_Float16*)alloc(nX * 2);

    const int thr = 256;
    // f32 -> f16 conversions
    cvt_f32_to_f16<<<(int)((nX + thr - 1) / thr), thr, 0, stream>>>(x, x_h, (int)nX);
    transpose_cvt_f16<<<(int)((nW + thr - 1) / thr), thr, 0, stream>>>(Wq, Wq_t, EMB, EMB);
    transpose_cvt_f16<<<(int)((nW + thr - 1) / thr), thr, 0, stream>>>(Wk, Wk_t, EMB, EMB);
    transpose_cvt_f16<<<(int)((nW + thr - 1) / thr), thr, 0, stream>>>(Wv, Wv_t, EMB, EMB);
    transpose_cvt_f16<<<(int)((nW + thr - 1) / thr), thr, 0, stream>>>(Wo, Wo_t, EMB, EMB);

    // QKV projections (f16 out)
    dim3 gg(EMB / 64, NTOK / 128);   // (16, 32)
    gemm_bias_wmma<<<gg, 256, 0, stream>>>(x_h, Wq_t, bq, q_h, nullptr,
                                           NTOK, EMB, EMB, 0);
    gemm_bias_wmma<<<gg, 256, 0, stream>>>(x_h, Wk_t, bk, k_h, nullptr,
                                           NTOK, EMB, EMB, 0);
    gemm_bias_wmma<<<gg, 256, 0, stream>>>(x_h, Wv_t, bv, v_h, nullptr,
                                           NTOK, EMB, EMB, 0);

    // V -> V^T per head
    transpose_v_kernel<<<(int)((nX + thr - 1) / thr), thr, 0, stream>>>(v_h, v_t);

    // flash attention: B*H*(S/64) = 1024 blocks of 128 threads
    flash_attn_wmma<<<BATCH * NHEAD * (SEQ / 64), 128, 0, stream>>>(q_h, k_h, v_t, attn_h);

    // output projection (f32 out -> d_out)
    gemm_bias_wmma<<<gg, 256, 0, stream>>>(attn_h, Wo_t, bo, nullptr, out,
                                           NTOK, EMB, EMB, 1);
}
